// AttentionHead_79740362818196
// MI455X (gfx1250) — matrix operand order repaired and checked by reading, not compiled, not run on hardware
//
#include <hip/hip_runtime.h>
#include <hip/hip_bf16.h>

// ---------------------------------------------------------------------------
// AttentionHead for MI455X (gfx1250, wave32, WMMA bf16 16x16x32, f32 accum)
//   B=4, T=4096, E=1024, d=64, causal softmax attention.
//   Round 3: 64-key chunks (16 WMMAs/step, fixed softmax-reduction cost
//   amortized 2x), grouped fragment loads for load/WMMA overlap, 1/sqrt(d)
//   folded into Wq. TDM double-buffered K/V staging kept.
// ---------------------------------------------------------------------------

typedef __bf16 bf16_t;
typedef __attribute__((ext_vector_type(16))) __bf16 v16bf;
typedef __attribute__((ext_vector_type(8)))  __bf16 v8bf;
typedef __attribute__((ext_vector_type(8)))  float  v8f;
typedef __attribute__((ext_vector_type(4)))  unsigned int v4u;
typedef __attribute__((ext_vector_type(4)))  int v4i;
typedef __attribute__((ext_vector_type(8)))  int v8i;

#define ATT_B 4
#define ATT_T 4096
#define ATT_E 1024
#define ATT_D 64

union V16 {
  v16bf v;
  v8bf  h[2];
};

__device__ __forceinline__ v8f wmma_bf16(v16bf a, v16bf b, v8f c) {
  return __builtin_amdgcn_wmma_f32_16x16x32_bf16(
      false, a, false, b, (short)0, c, false, false);
}

// A fragment (16x32), source row-major bf16 with leading dim ld.
__device__ __forceinline__ v16bf load_a_bf16(const bf16_t* tile, int ld) {
  const int lane = threadIdx.x & 31;
  const int m    = lane & 15;
  const int koff = (lane < 16) ? 0 : 8;
  const bf16_t* p = tile + (size_t)m * ld + koff;
  V16 r;
  r.h[0] = *(const v8bf*)(p);
  r.h[1] = *(const v8bf*)(p + 16);
  return r.v;
}

// A fragment from f32 row-major source, converted to bf16 in-register.
__device__ __forceinline__ v16bf load_a_f32(const float* tile, int ld) {
  const int lane = threadIdx.x & 31;
  const int m    = lane & 15;
  const int koff = (lane < 16) ? 0 : 8;
  const float* p = tile + (size_t)m * ld + koff;
  v8f f0 = *(const v8f*)(p);
  v8f f1 = *(const v8f*)(p + 16);
  V16 r;
#pragma unroll
  for (int j = 0; j < 8; ++j) {
    r.h[0][j] = (bf16_t)f0[j];
    r.h[1][j] = (bf16_t)f1[j];
  }
  return r.v;
}

// B fragment (logical B[K=0..31][N=0..15]) from storage laid out [N][K].
__device__ __forceinline__ v16bf load_b_bf16(const bf16_t* base, int ld) {
  const int lane = threadIdx.x & 31;
  const int nn   = lane & 15;
  const int krow = (lane < 16) ? 0 : 8;
  const bf16_t* p = base + (size_t)nn * ld + krow;
  V16 r;
  r.h[0] = *(const v8bf*)(p);
  r.h[1] = *(const v8bf*)(p + 16);
  return r.v;
}

// ---------------------------------------------------------------------------
// TDM: 2D bf16 tile load global->LDS (dims/strides in 2-byte elements).
// ---------------------------------------------------------------------------
__device__ __forceinline__ void tdm_load_2d_bf16(const void* gaddr, unsigned lds_off,
                                                 unsigned tile_d0, unsigned tile_d1,
                                                 unsigned tensor_d0, unsigned tensor_d1,
                                                 unsigned long long d0_stride) {
  const unsigned long long ga = (unsigned long long)(uintptr_t)gaddr;
  union { v4u v; unsigned long long q[2]; } g0;
  g0.q[0] = 1ull | ((unsigned long long)lds_off << 32);
  g0.q[1] = (ga & 0x01FFFFFFFFFFFFFFull) | (2ull << 62);  // type=2
  union { v8i v; unsigned long long q[4]; } g1;
  const unsigned long long data_size = 1;  // 2-byte elements
  g1.q[0] = (data_size << 16) | ((unsigned long long)(tensor_d0 & 0xFFFFu) << 48);
  g1.q[1] = ((unsigned long long)tensor_d0 >> 16) |
            ((unsigned long long)tensor_d1 << 16) |
            ((unsigned long long)(tile_d0 & 0xFFFFu) << 48);
  g1.q[2] = (unsigned long long)(tile_d1 & 0xFFFFu) |
            ((d0_stride & 0xFFFFFFFFull) << 32);
  g1.q[3] = (d0_stride >> 32) & 0xFFFFull;
  const v4i z4 = {0, 0, 0, 0};
#if defined(__clang_major__) && __clang_major__ >= 23
  const v8i z8 = {0, 0, 0, 0, 0, 0, 0, 0};
  __builtin_amdgcn_tensor_load_to_lds(g0.v, g1.v, z4, z4, z8, 0);
#else
  __builtin_amdgcn_tensor_load_to_lds(g0.v, g1.v, z4, z4, 0);
#endif
}

// ---------------------------------------------------------------------------
// Kernel 0: transpose + bf16-convert [1024][64] weight into [64][1024],
// optionally folding a scalar (1/sqrt(d) for Wq).
// ---------------------------------------------------------------------------
__global__ void att_wprep(const float* __restrict__ W, bf16_t* __restrict__ Wt,
                          float mul) {
  int idx = blockIdx.x * blockDim.x + threadIdx.x;
  if (idx < ATT_E * ATT_D) {
    int e = idx >> 6;
    int c = idx & 63;
    Wt[(size_t)c * ATT_E + e] = (bf16_t)(W[idx] * mul);
  }
}

// ---------------------------------------------------------------------------
// Kernel 1: projection Y = X[16384,1024] * W[1024,64], bf16 output.
// HBM-bound (streams 64MB f32 per call).
// ---------------------------------------------------------------------------
template <bool TRANSPOSED>
__global__ __launch_bounds__(256) void att_proj(const float* __restrict__ X,
                                                const bf16_t* __restrict__ Wt,
                                                bf16_t* __restrict__ Y) {
  const int lane = threadIdx.x & 31;
  const int wave = threadIdx.x >> 5;
  const int row0 = blockIdx.x * 128 + wave * 16;

  v8f acc[4] = {v8f{}, v8f{}, v8f{}, v8f{}};

  const float* Xrow = X + (size_t)row0 * ATT_E;
  for (int e0 = 0; e0 < ATT_E; e0 += 32) {
    v16bf a = load_a_f32(Xrow + e0, ATT_E);
    v16bf b0 = load_b_bf16(Wt + (size_t)0  * ATT_E + e0, ATT_E);
    v16bf b1 = load_b_bf16(Wt + (size_t)16 * ATT_E + e0, ATT_E);
    v16bf b2 = load_b_bf16(Wt + (size_t)32 * ATT_E + e0, ATT_E);
    v16bf b3 = load_b_bf16(Wt + (size_t)48 * ATT_E + e0, ATT_E);
    acc[0] = wmma_bf16(a, b0, acc[0]);
    acc[1] = wmma_bf16(a, b1, acc[1]);
    acc[2] = wmma_bf16(a, b2, acc[2]);
    acc[3] = wmma_bf16(a, b3, acc[3]);
  }

  const int nn = lane & 15;
  const int hi = lane >> 4;
  if (!TRANSPOSED) {
#pragma unroll
    for (int nt = 0; nt < 4; ++nt)
#pragma unroll
      for (int r = 0; r < 8; ++r)
        Y[(size_t)(row0 + r + 8 * hi) * ATT_D + nt * 16 + nn] = (bf16_t)acc[nt][r];
  } else {
    const int bidx = row0 / ATT_T;
    const int t0   = row0 % ATT_T;
#pragma unroll
    for (int nt = 0; nt < 4; ++nt)
#pragma unroll
      for (int r = 0; r < 8; ++r)
        Y[((size_t)bidx * ATT_D + nt * 16 + nn) * ATT_T + t0 + r + 8 * hi] =
            (bf16_t)acc[nt][r];
  }
}

// ---------------------------------------------------------------------------
// Kernel 2: causal flash attention, 64-key chunks, TDM double-buffered K/V.
// Block = 8 waves = 8 consecutive 16-query tiles (128 queries) of one batch.
// ---------------------------------------------------------------------------
__global__ __launch_bounds__(256) void att_flash(const bf16_t* __restrict__ qb,
                                                 const bf16_t* __restrict__ kb,
                                                 const bf16_t* __restrict__ vT,
                                                 float* __restrict__ out) {
  const int lane = threadIdx.x & 31;
  const int wave = threadIdx.x >> 5;

  __shared__ __align__(32) bf16_t s_k[2][64 * 64];    // [buf][key][d]   8KB each
  __shared__ __align__(32) bf16_t s_v[2][64 * 64];    // [buf][d][key]   8KB each
  __shared__ __align__(32) bf16_t lds_p[8][16 * 64];  // per-wave P tile, 2KB each
  bf16_t* myp = lds_p[wave];

  const int tile = blockIdx.x * 8 + wave;       // 0..1023
  const int b    = tile >> 8;                   // 256 tiles per batch
  const int q0   = (tile & 255) * 16;
  const int blk_q0 = ((blockIdx.x * 8) & 255) * 16;
  const int kend   = blk_q0 + 128;              // multiple of 128 -> 64 divides it

  const bf16_t* qbase = qb + ((size_t)b * ATT_T + q0) * ATT_D;
  const bf16_t* kbase = kb + (size_t)b * ATT_T * ATT_D;
  const bf16_t* vbase = vT + (size_t)b * ATT_D * ATT_T;

  const v16bf aq0 = load_a_bf16(qbase + 0, ATT_D);
  const v16bf aq1 = load_a_bf16(qbase + 32, ATT_D);

  v8f acc0{}, acc1{}, acc2{}, acc3{};
  float m[8], l[8];
#pragma unroll
  for (int r = 0; r < 8; ++r) { m[r] = -1e30f; l[r] = 0.0f; }

  const int nn = lane & 15;
  const int hi = lane >> 4;

  // ---- preload chunk 0 into buffer 0 (wave 0 drives the TDM)
  if (wave == 0) {
    tdm_load_2d_bf16(kbase, (unsigned)(uintptr_t)&s_k[0][0],
                     64, 64, 64, ATT_T, 64);
    tdm_load_2d_bf16(vbase, (unsigned)(uintptr_t)&s_v[0][0],
                     64, 64, ATT_T, 64, ATT_T);
    __builtin_amdgcn_s_wait_tensorcnt(0);
  }
  __syncthreads();

  for (int k0 = 0; k0 < kend; k0 += 64) {
    const int cur = (k0 >> 6) & 1;
    const int nxt = cur ^ 1;

    const bool pf = (k0 + 64 < kend);
    if (wave == 0 && pf) {
      tdm_load_2d_bf16(kbase + (size_t)(k0 + 64) * ATT_D,
                       (unsigned)(uintptr_t)&s_k[nxt][0],
                       64, 64, 64, ATT_T, 64);
      tdm_load_2d_bf16(vbase + (k0 + 64),
                       (unsigned)(uintptr_t)&s_v[nxt][0],
                       64, 64, ATT_T, 64, ATT_T);
    }

    if (k0 <= q0 + 15) {  // causal: this wave still has unmasked keys here
      const bf16_t* kt = &s_k[cur][0];   // [key][64]
      const bf16_t* vt = &s_v[cur][0];   // [d][64]

      // ---- scores for keys [k0, k0+64): 4 C tiles; grouped loads for overlap
      v8f s[4] = {v8f{}, v8f{}, v8f{}, v8f{}};
      {
        v16bf bk0 = load_b_bf16(kt + 0 * 64 + 0, 64);
        v16bf bk1 = load_b_bf16(kt + 16 * 64 + 0, 64);
        v16bf bk2 = load_b_bf16(kt + 32 * 64 + 0, 64);
        v16bf bk3 = load_b_bf16(kt + 48 * 64 + 0, 64);
        s[0] = wmma_bf16(aq0, bk0, s[0]);
        s[1] = wmma_bf16(aq0, bk1, s[1]);
        s[2] = wmma_bf16(aq0, bk2, s[2]);
        s[3] = wmma_bf16(aq0, bk3, s[3]);
        bk0 = load_b_bf16(kt + 0 * 64 + 32, 64);
        bk1 = load_b_bf16(kt + 16 * 64 + 32, 64);
        bk2 = load_b_bf16(kt + 32 * 64 + 32, 64);
        bk3 = load_b_bf16(kt + 48 * 64 + 32, 64);
        s[0] = wmma_bf16(aq1, bk0, s[0]);
        s[1] = wmma_bf16(aq1, bk1, s[1]);
        s[2] = wmma_bf16(aq1, bk2, s[2]);
        s[3] = wmma_bf16(aq1, bk3, s[3]);
      }

      // ---- causal mask (scale already folded into Wq) + online softmax
      float p[4][8], corr[8];
#pragma unroll
      for (int j = 0; j < 4; ++j)
#pragma unroll
        for (int r = 0; r < 8; ++r) {
          const int qg = q0 + r + 8 * hi;
          float x = s[j][r];
          if (k0 + j * 16 + nn > qg) x = -1e30f;
          p[j][r] = x;
        }
#pragma unroll
      for (int r = 0; r < 8; ++r) {
        float t = fmaxf(fmaxf(p[0][r], p[1][r]), fmaxf(p[2][r], p[3][r]));
#pragma unroll
        for (int off = 8; off > 0; off >>= 1)
          t = fmaxf(t, __shfl_xor(t, off, 16));
        const float mnew = fmaxf(m[r], t);
        corr[r] = __expf(m[r] - mnew);
        m[r] = mnew;
        float rs = 0.0f;
#pragma unroll
        for (int j = 0; j < 4; ++j) {
          const float e = __expf(p[j][r] - mnew);
          p[j][r] = e;
          rs += e;
        }
#pragma unroll
        for (int off = 8; off > 0; off >>= 1)
          rs += __shfl_xor(rs, off, 16);
        l[r] = l[r] * corr[r] + rs;
      }
#pragma unroll
      for (int r = 0; r < 8; ++r) {
        acc0[r] *= corr[r];
        acc1[r] *= corr[r];
        acc2[r] *= corr[r];
        acc3[r] *= corr[r];
      }

      // ---- C-layout -> A-layout via per-wave LDS tile (row-major [16][64])
#pragma unroll
      for (int r = 0; r < 8; ++r) {
        const int row = r + 8 * hi;
#pragma unroll
        for (int j = 0; j < 4; ++j)
          myp[row * 64 + j * 16 + nn] = (bf16_t)p[j][r];
      }
      asm volatile("s_wait_dscnt 0x0" ::: "memory");
      const v16bf ap0 = load_a_bf16(myp, 64);        // keys k0   .. k0+31
      const v16bf ap1 = load_a_bf16(myp + 32, 64);   // keys k0+32.. k0+63

      // ---- PV: grouped B loads, 8 WMMAs
      {
        v16bf bv0 = load_b_bf16(vt + 0  * 64 + 0, 64);
        v16bf bv1 = load_b_bf16(vt + 16 * 64 + 0, 64);
        v16bf bv2 = load_b_bf16(vt + 32 * 64 + 0, 64);
        v16bf bv3 = load_b_bf16(vt + 48 * 64 + 0, 64);
        acc0 = wmma_bf16(ap0, bv0, acc0);
        acc1 = wmma_bf16(ap0, bv1, acc1);
        acc2 = wmma_bf16(ap0, bv2, acc2);
        acc3 = wmma_bf16(ap0, bv3, acc3);
        bv0 = load_b_bf16(vt + 0  * 64 + 32, 64);
        bv1 = load_b_bf16(vt + 16 * 64 + 32, 64);
        bv2 = load_b_bf16(vt + 32 * 64 + 32, 64);
        bv3 = load_b_bf16(vt + 48 * 64 + 32, 64);
        acc0 = wmma_bf16(ap1, bv0, acc0);
        acc1 = wmma_bf16(ap1, bv1, acc1);
        acc2 = wmma_bf16(ap1, bv2, acc2);
        acc3 = wmma_bf16(ap1, bv3, acc3);
      }
    }

    if (wave == 0 && pf) __builtin_amdgcn_s_wait_tensorcnt(0);
    __syncthreads();
  }

  // ---- epilogue: out[b][q][d] = acc / l
  float* op = out + ((size_t)b * ATT_T + q0) * ATT_D;
#pragma unroll
  for (int r = 0; r < 8; ++r) {
    const float inv = 1.0f / l[r];
    const int row = r + 8 * hi;
    op[(size_t)row * ATT_D + 0  + nn] = acc0[r] * inv;
    op[(size_t)row * ATT_D + 16 + nn] = acc1[r] * inv;
    op[(size_t)row * ATT_D + 32 + nn] = acc2[r] * inv;
    op[(size_t)row * ATT_D + 48 + nn] = acc3[r] * inv;
  }
}

// ---------------------------------------------------------------------------
// Launch
// ---------------------------------------------------------------------------
extern "C" void kernel_launch(void* const* d_in, const int* in_sizes, int n_in,
                              void* d_out, int out_size, void* d_ws, size_t ws_size,
                              hipStream_t stream) {
  (void)in_sizes; (void)n_in; (void)out_size; (void)ws_size;
  const float* K  = (const float*)d_in[0];
  const float* Q  = (const float*)d_in[1];
  const float* V  = (const float*)d_in[2];
  const float* Wk = (const float*)d_in[3];
  const float* Wq = (const float*)d_in[4];
  const float* Wv = (const float*)d_in[5];
  float* out = (float*)d_out;

  const size_t rows = (size_t)ATT_B * ATT_T;          // 16384
  bf16_t* qb  = (bf16_t*)d_ws;                        // [rows][64]
  bf16_t* kb  = qb + rows * ATT_D;                    // [rows][64]
  bf16_t* vT  = kb + rows * ATT_D;                    // [B][64][T]
  bf16_t* WqT = vT + rows * ATT_D;                    // [64][1024]
  bf16_t* WkT = WqT + (size_t)ATT_D * ATT_E;
  bf16_t* WvT = WkT + (size_t)ATT_D * ATT_E;

  // 1/sqrt(64) folded into Wq.
  att_wprep<<<(ATT_E * ATT_D + 255) / 256, 256, 0, stream>>>(Wq, WqT, 0.125f);
  att_wprep<<<(ATT_E * ATT_D + 255) / 256, 256, 0, stream>>>(Wk, WkT, 1.0f);
  att_wprep<<<(ATT_E * ATT_D + 255) / 256, 256, 0, stream>>>(Wv, WvT, 1.0f);

  att_proj<false><<<128, 256, 0, stream>>>(Q, WqT, qb);
  att_proj<false><<<128, 256, 0, stream>>>(K, WkT, kb);
  att_proj<true ><<<128, 256, 0, stream>>>(V, WvT, vT);

  att_flash<<<128, 256, 0, stream>>>(qb, kb, vT, out);
}
